// LocalTransformer_60395830116430
// MI455X (gfx1250) — compile-verified, matrix operand order checked by reading, not run on hardware
//
#include <hip/hip_runtime.h>
#include <math.h>

// ---------------- types ----------------
typedef __attribute__((ext_vector_type(16))) __bf16 v16bf;
typedef __attribute__((ext_vector_type(8)))  float  v8f;
typedef __attribute__((ext_vector_type(4)))  unsigned int uint4v;

static constexpr int Bc    = 2;
static constexpr int Nc    = 4096;
static constexpr int DIMc  = 1024;
static constexpr int HEADS = 16;
static constexpr int DH    = 64;
static constexpr int Wc    = 128;
static constexpr int NWc   = 32;
static constexpr int Jc    = 384;          // 3*W keys per window
static constexpr int FFI   = 2730;         // FF_INNER
static constexpr int ROWS  = Bc * Nc;      // 8192 tokens

__device__ inline unsigned short f32_to_bf16(float f) {
  unsigned int u = __float_as_uint(f);
  unsigned int r = u + 0x7FFFu + ((u >> 16) & 1u);
  return (unsigned short)(r >> 16);
}

__device__ inline v8f zero8() { v8f z; for (int i = 0; i < 8; ++i) z[i] = 0.f; return z; }

__device__ inline v8f wmma_bf16(v16bf a, v16bf b, v8f c) {
  return __builtin_amdgcn_wmma_f32_16x16x32_bf16(false, a, false, b, (short)0, c, false, false);
}

// Load a 16x32 bf16 fragment (A or B-transposed layout) from LDS.
// Element e maps to K = (e>>3)*16 + ((lane>=16)?8:0) + (e&7)  -> two 16B reads.
__device__ inline v16bf frag16(const unsigned short* rowptr, int lane) {
  int h8 = (lane >> 4) << 3; // 0 or 8
  union { uint4v q[2]; v16bf v; } f;
  f.q[0] = *(const uint4v*)(rowptr + h8);
  f.q[1] = *(const uint4v*)(rowptr + 16 + h8);
  return f.v;
}

// Async global->LDS copy of 16 bytes per lane (ASYNCcnt-tracked, CDNA5 TDM-lite path).
__device__ inline void async_ld_b128(unsigned lds_byte_off, const void* gptr) {
  asm volatile("global_load_async_to_lds_b128 %0, %1, off"
               :
               : "v"(lds_byte_off), "v"((unsigned long long)(size_t)gptr)
               : "memory");
}
__device__ inline void wait_async0() {
  asm volatile("s_wait_asynccnt 0x0" ::: "memory");
}

// ---------------- LayerNorm -> bf16 ----------------
__global__ __launch_bounds__(256)
void ln_kernel(const float* __restrict__ x, const float* __restrict__ g,
               const float* __restrict__ bb, unsigned short* __restrict__ out) {
  __shared__ float red[8];
  int row = blockIdx.x, tid = threadIdx.x;
  const float* xr = x + (size_t)row * DIMc;
  float v[4]; float s = 0.f;
  for (int i = 0; i < 4; ++i) { v[i] = xr[tid + 256 * i]; s += v[i]; }
  for (int m = 16; m; m >>= 1) s += __shfl_xor(s, m, 32);
  if ((tid & 31) == 0) red[tid >> 5] = s;
  __syncthreads();
  s = 0.f;
  for (int i = 0; i < 8; ++i) s += red[i];
  float mu = s * (1.f / DIMc);
  __syncthreads();
  float s2 = 0.f;
  for (int i = 0; i < 4; ++i) { float d = v[i] - mu; s2 += d * d; }
  for (int m = 16; m; m >>= 1) s2 += __shfl_xor(s2, m, 32);
  if ((tid & 31) == 0) red[tid >> 5] = s2;
  __syncthreads();
  s2 = 0.f;
  for (int i = 0; i < 8; ++i) s2 += red[i];
  float rs = rsqrtf(s2 * (1.f / DIMc) + 1e-5f);
  unsigned short* orow = out + (size_t)row * DIMc;
  for (int i = 0; i < 4; ++i) {
    int c = tid + 256 * i;
    orow[c] = f32_to_bf16((v[i] - mu) * rs * g[c] + bb[c]);
  }
}

// ---------------- f32 -> bf16 convert ----------------
__global__ __launch_bounds__(256)
void cvt_kernel(const float* __restrict__ src, unsigned short* __restrict__ dst, int n) {
  int i = blockIdx.x * 256 + threadIdx.x;
  if (i < n) dst[i] = f32_to_bf16(src[i]);
}

// ---------------- bf16 GEMM: C[M,N](f32) = A[M,K] @ B[K,N] (+ res) ----------------
// Double-buffered LDS pipeline: A tile via GLOBAL_LOAD_ASYNC_TO_LDS_B128 (ASYNCcnt),
// B tile transposed via VGPR path, 8 WMMAs per wave per 32-K step.
__global__ __launch_bounds__(256)
void gemm_kernel(const unsigned short* __restrict__ A,
                 const unsigned short* __restrict__ Bw,
                 float* __restrict__ C,
                 const float* __restrict__ res,
                 int M, int N, int K, int addres) {
  // [buf 0 | buf 1], each buf = As(128x32) then Bt(128x32); LDS base of smem == 0.
  __shared__ __align__(16) unsigned short smem[2 * 8192];
  int tid = threadIdx.x, lane = tid & 31, wid = tid >> 5;
  int n0 = blockIdx.x * 128, m0 = blockIdx.y * 128;
  int wm = wid & 3, wn = wid >> 2;   // 4 waves in M (32 rows each), 2 in N (64 cols each)
  int rsel = lane & 15;
  bool k8 = ((K & 7) == 0);
  bool n8 = ((N & 7) == 0);
  int nsteps = (K + 31) / 32;

  v8f acc[2][4];
  for (int s = 0; s < 2; ++s)
    for (int t = 0; t < 4; ++t) acc[s][t] = zero8();

  auto load_tiles = [&](int k0, int buf) {
    unsigned short* As = smem + buf * 8192;
    unsigned short* Bt = As + 4096;
    // ---- A tile: 128 rows x 32 k, row-major ----
    if (k8 && (k0 + 32 <= K)) {
      // pure async path: 2 x 16B per lane, no VGPR staging
      int idx = tid * 8;
      for (int it = 0; it < 2; ++it, idx += 2048) {
        int row = idx >> 5, ko = idx & 31;
        unsigned lds_off = (unsigned)((buf * 8192 + idx) * 2);
        async_ld_b128(lds_off, A + (size_t)(m0 + row) * K + k0 + ko);
      }
    } else {
      int r = tid >> 2, ko = (tid & 3) * 8;
      for (int rr = r; rr < 128; rr += 64) {
        const unsigned short* src = A + (size_t)(m0 + rr) * K + k0 + ko;
        union { uint4v q; unsigned short u[8]; } val;
        for (int e = 0; e < 8; ++e) val.u[e] = (k0 + ko + e < K) ? src[e] : (unsigned short)0;
        *(uint4v*)(As + rr * 32 + ko) = val.q;
      }
    }
    // ---- B tile: 32 k x 128 n, stored transposed Bt[n][k] ----
    {
      int kr = tid >> 3;            // 0..31
      int nb = (tid & 7) * 16;      // 0..112
      const unsigned short* src = Bw + (size_t)(k0 + kr) * N + n0 + nb;
      bool krok = (k0 + kr) < K;
      if (krok && n8 && (n0 + nb + 16 <= N)) {
        union { uint4v q; unsigned short u[8]; } a0, a1;
        a0.q = *(const uint4v*)src;
        a1.q = *(const uint4v*)(src + 8);
        for (int c = 0; c < 8; ++c) Bt[(nb + c) * 32 + kr] = a0.u[c];
        for (int c = 0; c < 8; ++c) Bt[(nb + 8 + c) * 32 + kr] = a1.u[c];
      } else {
        for (int c = 0; c < 16; ++c) {
          unsigned short v = 0;
          if (krok && (n0 + nb + c) < N) v = src[c];
          Bt[(nb + c) * 32 + kr] = v;
        }
      }
    }
  };

  // prologue: fill buffer 0
  load_tiles(0, 0);
  wait_async0();
  __syncthreads();

  for (int step = 0; step < nsteps; ++step) {
    int cur = step & 1;
    // overlap: start filling the other buffer while we compute from this one
    if (step + 1 < nsteps) load_tiles((step + 1) * 32, cur ^ 1);

    const unsigned short* As = smem + cur * 8192;
    const unsigned short* Bt = As + 4096;
    v16bf af[2], bfr[4];
    af[0] = frag16(As + (wm * 32 + 0  + rsel) * 32, lane);
    af[1] = frag16(As + (wm * 32 + 16 + rsel) * 32, lane);
    for (int t = 0; t < 4; ++t)
      bfr[t] = frag16(Bt + (wn * 64 + t * 16 + rsel) * 32, lane);
    for (int s = 0; s < 2; ++s)
      for (int t = 0; t < 4; ++t)
        acc[s][t] = wmma_bf16(af[s], bfr[t], acc[s][t]);

    wait_async0();     // next buffer's async fills complete
    __syncthreads();   // + everyone done reading current buffer
  }

  // ---- store (C/D layout: lane<16 -> M=r, lane>=16 -> M=8+r; N=lane&15) ----
  int mrow = (lane >> 4) << 3;
  for (int s = 0; s < 2; ++s)
    for (int t = 0; t < 4; ++t) {
      int gn = n0 + wn * 64 + t * 16 + rsel;
      if (gn >= N) continue;
      for (int r = 0; r < 8; ++r) {
        int gm = m0 + wm * 32 + s * 16 + r + mrow;
        size_t idx = (size_t)gm * N + gn;
        float v = acc[s][t][r];
        if (addres) v += res[idx];
        C[idx] = v;
      }
    }
}

// ---------------- windowed attention ----------------
// grid (32 heads*batch, 32 windows), 256 threads. qkv: [8192][3072] f32.
__global__ __launch_bounds__(256)
void attn_kernel(const float* __restrict__ qkv, const float* __restrict__ qs,
                 const float* __restrict__ ks, unsigned short* __restrict__ ob) {
  extern __shared__ __align__(16) char smem_raw[];
  unsigned short* bq    = (unsigned short*)smem_raw;   // 128*64
  unsigned short* bk    = bq  + 128 * 64;              // 384*64  [j][d]
  unsigned short* bvT   = bk  + 384 * 64;              // 64*384  [d][j]
  unsigned short* stage = bvT + 64 * 384;              // 8 waves * 16*32

  int bh = blockIdx.x, w = blockIdx.y;
  int batch = bh >> 4, h = bh & 15;
  int tid = threadIdx.x, lane = tid & 31, wid = tid >> 5;
  int rsel = lane & 15;

  // ---- Q: l2norm * q_scale * 8, RoPE/xpos at t = 256 + i ----
  for (int rr = wid; rr < 128; rr += 8) {
    int tok = batch * Nc + w * Wc + rr;
    const float* qrow = qkv + (size_t)tok * 3072 + h * 64;
    float x1 = qrow[lane], x2 = qrow[lane + 32];
    float ss = x1 * x1 + x2 * x2;
    for (int m = 16; m; m >>= 1) ss += __shfl_xor(ss, m, 32);
    float inv = 1.f / fmaxf(sqrtf(ss), 1e-12f);
    x1 = x1 * inv * qs[lane] * 8.0f;
    x2 = x2 * inv * qs[lane + 32] * 8.0f;
    float t = (float)(256 + rr);
    float fr = t * powf(10000.f, -(float)(2 * lane) / 64.f);
    float sn, cs; sincosf(fr, &sn, &cs);
    float xp = powf((2.f * lane + 25.6f) / 89.6f, (t - 192.f) / 64.f);
    cs *= xp; sn *= xp;
    bq[rr * 64 + lane]      = f32_to_bf16(x1 * cs - x2 * sn);
    bq[rr * 64 + lane + 32] = f32_to_bf16(x2 * cs + x1 * sn);
  }
  // ---- K: l2norm * k_scale, RoPE/xpos^-1 at t = j ----
  for (int j = wid; j < Jc; j += 8) {
    int kt = w * Wc - Wc + j;
    float o1 = 0.f, o2 = 0.f;
    float x1 = 0.f, x2 = 0.f;
    bool ok = (kt >= 0) && (kt < Nc);
    if (ok) {
      const float* krow = qkv + (size_t)(batch * Nc + kt) * 3072 + 1024 + h * 64;
      x1 = krow[lane]; x2 = krow[lane + 32];
    }
    float ss = x1 * x1 + x2 * x2;
    for (int m = 16; m; m >>= 1) ss += __shfl_xor(ss, m, 32);
    if (ok) {
      float inv = 1.f / fmaxf(sqrtf(ss), 1e-12f);
      x1 *= inv * ks[lane]; x2 *= inv * ks[lane + 32];
      float t = (float)j;
      float fr = t * powf(10000.f, -(float)(2 * lane) / 64.f);
      float sn, cs; sincosf(fr, &sn, &cs);
      float xp = powf((2.f * lane + 25.6f) / 89.6f, (t - 192.f) / 64.f);
      cs /= xp; sn /= xp;
      o1 = x1 * cs - x2 * sn;
      o2 = x2 * cs + x1 * sn;
    }
    bk[j * 64 + lane]      = f32_to_bf16(o1);
    bk[j * 64 + lane + 32] = f32_to_bf16(o2);
  }
  // ---- V: transposed [d][j] ----
  for (int j = wid; j < Jc; j += 8) {
    int kt = w * Wc - Wc + j;
    float x1 = 0.f, x2 = 0.f;
    if (kt >= 0 && kt < Nc) {
      const float* vrow = qkv + (size_t)(batch * Nc + kt) * 3072 + 2048 + h * 64;
      x1 = vrow[lane]; x2 = vrow[lane + 32];
    }
    bvT[lane * 384 + j]        = f32_to_bf16(x1);
    bvT[(lane + 32) * 384 + j] = f32_to_bf16(x2);
  }
  __syncthreads();

  // ---- sim = bq @ bk^T : per-wave 16-row strip, 24 col tiles ----
  int strip = wid * 16;
  v16bf aq0 = frag16(bq + (strip + rsel) * 64, lane);
  v16bf aq1 = frag16(bq + (strip + rsel) * 64 + 32, lane);
  v8f acc[24];
  for (int t = 0; t < 24; ++t) acc[t] = zero8();
  for (int jt = 0; jt < 24; ++jt) {
    const unsigned short* krow = bk + (jt * 16 + rsel) * 64;
    v16bf b0 = frag16(krow, lane);
    v16bf b1 = frag16(krow + 32, lane);
    acc[jt] = wmma_bf16(aq0, b0, acc[jt]);
    acc[jt] = wmma_bf16(aq1, b1, acc[jt]);
  }

  // ---- mask + softmax (rows live in lane halves: M = r + 8*(lane>=16)) ----
  int mrow = (lane >> 4) << 3;
  for (int r = 0; r < 8; ++r) {
    int m = strip + r + mrow;
    float mx = -1e30f;
    for (int jt = 0; jt < 24; ++jt) {
      int j = jt * 16 + rsel;
      int kt = w * Wc - Wc + j;
      int d = m + 128 - j;
      bool val = (kt >= 0) && (kt < Nc) && (d >= -128) && (d <= 128);
      float vv = val ? acc[jt][r] : -1e30f;
      acc[jt][r] = vv;
      mx = fmaxf(mx, vv);
    }
    for (int mk = 8; mk; mk >>= 1) mx = fmaxf(mx, __shfl_xor(mx, mk, 32));
    float sum = 0.f;
    for (int jt = 0; jt < 24; ++jt) {
      float p = expf(acc[jt][r] - mx);
      acc[jt][r] = p;
      sum += p;
    }
    for (int mk = 8; mk; mk >>= 1) sum += __shfl_xor(sum, mk, 32);
    float rinv = 1.f / sum;
    for (int jt = 0; jt < 24; ++jt) acc[jt][r] *= rinv;
  }

  // ---- o = attn @ v : stage probs per-wave, 12 k-steps of 32 ----
  unsigned short* st = stage + wid * (16 * 32);
  v8f oacc[4];
  for (int t = 0; t < 4; ++t) oacc[t] = zero8();
  for (int jt2 = 0; jt2 < 12; ++jt2) {
    for (int hlf = 0; hlf < 2; ++hlf) {
      int t = 2 * jt2 + hlf;
      for (int r = 0; r < 8; ++r)
        st[(r + mrow) * 32 + rsel + 16 * hlf] = f32_to_bf16(acc[t][r]);
    }
    v16bf ap = frag16(st + rsel * 32, lane);
    for (int nt = 0; nt < 4; ++nt) {
      const unsigned short* vrow = bvT + (nt * 16 + rsel) * 384 + jt2 * 32;
      oacc[nt] = wmma_bf16(ap, frag16(vrow, lane), oacc[nt]);
    }
  }

  // ---- write o as bf16, (B,N,HEADS*DH) layout ----
  for (int nt = 0; nt < 4; ++nt)
    for (int r = 0; r < 8; ++r) {
      int m = strip + r + mrow;
      int tok = batch * Nc + w * Wc + m;
      int col = h * 64 + nt * 16 + rsel;
      ob[(size_t)tok * 1024 + col] = f32_to_bf16(oacc[nt][r]);
    }
}

// ---------------- GEGLU: f = u * gelu(gate) ----------------
__global__ __launch_bounds__(256)
void geglu_kernel(const float* __restrict__ ug, unsigned short* __restrict__ f) {
  long long i = (long long)blockIdx.x * 256 + threadIdx.x;
  const long long total = (long long)ROWS * FFI;
  if (i >= total) return;
  long long row = i / FFI;
  int c = (int)(i - row * FFI);
  const float* rp = ug + row * (2LL * FFI);
  float u = rp[c], g = rp[FFI + c];
  float ge = 0.5f * g * (1.f + erff(g * 0.70710678118654752f));
  f[i] = f32_to_bf16(u * ge);
}

// ---------------- host orchestration ----------------
extern "C" void kernel_launch(void* const* d_in, const int* in_sizes, int n_in,
                              void* d_out, int out_size, void* d_ws, size_t ws_size,
                              hipStream_t stream) {
  const float* x_in    = (const float*)d_in[0];
  const float* ln1_g   = (const float*)d_in[1];
  const float* ln1_b   = (const float*)d_in[2];
  const float* w_qkv   = (const float*)d_in[3];
  const float* q_scale = (const float*)d_in[4];
  const float* k_scale = (const float*)d_in[5];
  const float* w_out   = (const float*)d_in[6];
  const float* ln2_g   = (const float*)d_in[7];
  const float* ln2_b   = (const float*)d_in[8];
  const float* w1      = (const float*)d_in[9];
  const float* w2      = (const float*)d_in[10];

  char* p = (char*)d_ws;
  auto take = [&](size_t bytes) {
    char* q = p;
    p += (bytes + 255) & ~(size_t)255;
    return (void*)q;
  };
  float*          xbuf = (float*)take((size_t)ROWS * DIMc * 4);       // residual stream
  float*          buf1 = (float*)take((size_t)ROWS * (2 * FFI) * 4);  // qkv / ugate
  unsigned short* hb   = (unsigned short*)take((size_t)ROWS * DIMc * 2);   // h bf16 / attn out bf16
  unsigned short* fb   = (unsigned short*)take((size_t)ROWS * FFI * 2);    // geglu out bf16
  unsigned short* wb   = (unsigned short*)take((size_t)DIMc * (2 * FFI) * 2); // weight bf16

  hipMemcpyAsync(xbuf, x_in, (size_t)ROWS * DIMc * 4, hipMemcpyDeviceToDevice, stream);

  const size_t attn_lds = (128 * 64 + 384 * 64 + 64 * 384 + 8 * 16 * 32) * sizeof(unsigned short);

  for (int L = 0; L < 8; ++L) {
    const float* g1 = ln1_g + L * DIMc;
    const float* b1 = ln1_b + L * DIMc;
    const float* g2 = ln2_g + L * DIMc;
    const float* b2 = ln2_b + L * DIMc;
    const float* wqkvL = w_qkv + (size_t)L * DIMc * 3072;
    const float* qsL   = q_scale + L * DH;
    const float* ksL   = k_scale + L * DH;
    const float* woL   = w_out + (size_t)L * 1024 * DIMc;
    const float* w1L   = w1 + (size_t)L * DIMc * (2 * FFI);
    const float* w2L   = w2 + (size_t)L * FFI * DIMc;

    // 1. ln1 -> hb (bf16)
    ln_kernel<<<ROWS, 256, 0, stream>>>(xbuf, g1, b1, hb);
    // 2. convert w_qkv
    {
      int n = DIMc * 3072;
      cvt_kernel<<<(n + 255) / 256, 256, 0, stream>>>(wqkvL, wb, n);
    }
    // 3. qkv = h @ w_qkv  -> buf1 (f32)
    gemm_kernel<<<dim3(3072 / 128, ROWS / 128), 256, 0, stream>>>(
        hb, wb, buf1, nullptr, ROWS, 3072, DIMc, 0);
    // 4. windowed attention -> hb (bf16 attention output)
    attn_kernel<<<dim3(Bc * HEADS, NWc), 256, attn_lds, stream>>>(buf1, qsL, ksL, hb);
    // 5. convert w_out
    {
      int n = 1024 * DIMc;
      cvt_kernel<<<(n + 255) / 256, 256, 0, stream>>>(woL, wb, n);
    }
    // 6. x += o @ w_out
    gemm_kernel<<<dim3(DIMc / 128, ROWS / 128), 256, 0, stream>>>(
        hb, wb, xbuf, xbuf, ROWS, DIMc, 1024, 1);
    // 7. ln2 -> hb
    ln_kernel<<<ROWS, 256, 0, stream>>>(xbuf, g2, b2, hb);
    // 8. convert w1
    {
      int n = DIMc * (2 * FFI);
      cvt_kernel<<<(n + 255) / 256, 256, 0, stream>>>(w1L, wb, n);
    }
    // 9. ugate = h @ w1 -> buf1 (N = 5460, tail-guarded)
    gemm_kernel<<<dim3((2 * FFI + 127) / 128, ROWS / 128), 256, 0, stream>>>(
        hb, wb, buf1, nullptr, ROWS, 2 * FFI, DIMc, 0);
    // 10. geglu -> fb (bf16)
    {
      long long total = (long long)ROWS * FFI;
      geglu_kernel<<<(unsigned)((total + 255) / 256), 256, 0, stream>>>(buf1, fb);
    }
    // 11. convert w2
    {
      int n = FFI * DIMc;
      cvt_kernel<<<(n + 255) / 256, 256, 0, stream>>>(w2L, wb, n);
    }
    // 12. x += f @ w2  (K = 2730, tail-guarded)
    gemm_kernel<<<dim3(DIMc / 128, ROWS / 128), 256, 0, stream>>>(
        fb, wb, xbuf, xbuf, ROWS, DIMc, FFI, 1);
  }

  hipMemcpyAsync(d_out, xbuf, (size_t)ROWS * DIMc * 4, hipMemcpyDeviceToDevice, stream);
}
